// TemporalAttention_59158879535134
// MI455X (gfx1250) — matrix operand order repaired and checked by reading, not compiled, not run on hardware
//
#include <hip/hip_runtime.h>
#include <hip/hip_bf16.h>

// ---------------------------------------------------------------------------
// TemporalAttention fused pipeline for gfx1250 (MI455X), wave32 + WMMA f16.
//   B=8 P=128 N=256 F=128 D_MODEL=128 H=8 dh=16, Din=256, M=B*P*N=262144
// Weights are pre-packed per launch into f16 fragment-major layout so the
// steady-state GEMM loops issue 1 v16h load per WMMA B operand.
// ---------------------------------------------------------------------------

#define B_   8
#define P_   128
#define N_   256
#define F_   128
#define D_   128     // D_MODEL
#define DIN_ 256
#define H_   8
#define DH_  16
#define NEG_INF_ (-65504.0f)

typedef __attribute__((ext_vector_type(16))) _Float16 v16h;
typedef __attribute__((ext_vector_type(8)))  float    v8f;

__device__ __forceinline__ v8f wmma_f16(const v16h& a, const v16h& b, const v8f& c) {
  // D(f32 16x16) = A(f16 16x32) * B(f16 32x16) + C
  return __builtin_amdgcn_wmma_f32_16x16x32_f16(
      /*neg_a=*/false, a, /*neg_b=*/false, b,
      /*c_mod=*/(short)0, c, /*reuse_a=*/false, /*reuse_b=*/false);
}

// A-matrix fragment (16x32 f16). lane L holds row M=L&15;
// lanes 0-15 hold K=0..15 sequentially, lanes 16-31 K=16..31.
template <typename F>
__device__ __forceinline__ v16h load_a_frag(int lane, F&& src) {
  v16h a;
  const int m  = lane & 15;
  const int kb = (lane >> 4) << 4;   // 0 or 16
#pragma unroll
  for (int v = 0; v < 8; ++v) {
    const int k0 = 2 * v + kb;
    a[2 * v]     = src(m, k0);
    a[2 * v + 1] = src(m, k0 + 1);
  }
  return a;
}

// B-matrix fragment (32x16 f16). lane L holds column N=L&15;
// lanes 0-15 hold K=0..15 sequentially, lanes 16-31 K=16..31.
template <typename F>
__device__ __forceinline__ v16h load_b_frag(int lane, F&& src) {
  v16h b;
  const int nn = lane & 15;
  const int kb = (lane >> 4) << 4;   // 0 or 16
#pragma unroll
  for (int v = 0; v < 8; ++v) {
    b[2 * v]     = src(kb + 2 * v, nn);
    b[2 * v + 1] = src(kb + 2 * v + 1, nn);
  }
  return b;
}
// C/D f32 16x16: element i of v8f -> row = i + 8*(lane>>4), col = lane&15.

// Packed-weight addressing: Wp[((c*ktiles + kt)*32 + lane)*16 .. +15] is the
// exact 16-half B-fragment for (col tile c, k tile kt, lane).
__device__ __forceinline__ const v16h* wp_frag(const _Float16* Wp, int ktiles,
                                               int c, int kt, int lane) {
  return reinterpret_cast<const v16h*>(
      Wp + (((size_t)c * ktiles + kt) * 32 + lane) * 16);
}

// ---------------------------------------------------------------------------
// Kernel 0: pack one f32 weight matrix (K x 128) into f16 fragment-major.
// grid = ktiles (K/32), block = 256 (wave w = col tile c).
// ---------------------------------------------------------------------------
__global__ __launch_bounds__(256) void pack_w_kernel(
    const float* __restrict__ W, _Float16* __restrict__ Wp, int ktiles)
{
  const int kt   = blockIdx.x;
  const int c    = threadIdx.x >> 5;
  const int lane = threadIdx.x & 31;
  const int nn   = lane & 15;
  const int kb   = (lane >> 4) << 4;
  v16h frag;
#pragma unroll
  for (int e = 0; e < 16; ++e) {
    const int kk = kb + e;
    frag[e] = (_Float16)W[(size_t)(kt * 32 + kk) * D_ + c * 16 + nn];
  }
  *reinterpret_cast<v16h*>(
      Wp + (((size_t)c * ktiles + kt) * 32 + lane) * 16) = frag;
}

// ---------------------------------------------------------------------------
// Kernel 1: q/k/v = relu(concat(X,STE) @ W + b), outputs f16 in head-major
// layout [b][h][n][p][dh]. grid = M/16 blocks of 256 (8 waves = 8 heads).
// ---------------------------------------------------------------------------
__global__ __launch_bounds__(256) void qkv_kernel(
    const float* __restrict__ X,  const float* __restrict__ STE,
    const _Float16* __restrict__ WqP, const float* __restrict__ bq,
    const _Float16* __restrict__ WkP, const float* __restrict__ bk,
    const _Float16* __restrict__ WvP, const float* __restrict__ bv,
    _Float16* __restrict__ qp, _Float16* __restrict__ kp,
    _Float16* __restrict__ vp)
{
  __shared__ _Float16 xs[16][DIN_ + 8];   // 16 rows x 256 (+pad), f16

  const int m0  = blockIdx.x * 16;
  const int tid = threadIdx.x;

  { // stage concat(X, STE) tile -> LDS (f32 -> f16)
    const int r  = tid >> 4;
    const int c0 = (tid & 15) * 16;
    const size_t mrow = (size_t)(m0 + r);
    const float* sx = X   + mrow * F_;
    const float* ss = STE + mrow * D_;
#pragma unroll
    for (int j = 0; j < 16; ++j) {
      const int c = c0 + j;
      const float v = (c < F_) ? sx[c] : ss[c - F_];
      xs[r][c] = (_Float16)v;
    }
  }
  __syncthreads();

  const int w    = tid >> 5;   // wave id = head id = output col tile
  const int lane = tid & 31;

  v16h afrag[8];
#pragma unroll
  for (int kt = 0; kt < 8; ++kt) {
    afrag[kt] = load_a_frag(lane, [&](int mm, int kk) -> _Float16 {
      return xs[mm][kt * 32 + kk];
    });
  }

  const int n0 = m0 % N_;
  const int bp = m0 / N_;
  const int p  = bp % P_;
  const int b  = bp / P_;

  const _Float16* Wpacks[3] = {WqP, WkP, WvP};
  const float* biases[3] = {bq, bk, bv};
  _Float16* outs[3] = {qp, kp, vp};

  const int col   = lane & 15;
  const int rbase = (lane >> 4) * 8;

#pragma unroll
  for (int mtx = 0; mtx < 3; ++mtx) {
    const _Float16* Wp = Wpacks[mtx];
    v8f acc = {};
#pragma unroll
    for (int kt = 0; kt < 8; ++kt) {
      const v16h bfr = *wp_frag(Wp, 8, w, kt, lane);   // one 32B load
      acc = wmma_f16(afrag[kt], bfr, acc);
    }
    const float bb = biases[mtx][w * 16 + col];
    _Float16* dst = outs[mtx];
#pragma unroll
    for (int i = 0; i < 8; ++i) {
      const int r = rbase + i;            // row within tile -> n = n0 + r
      float v = acc[i] + bb;
      v = fmaxf(v, 0.0f);
      const size_t o =
          (((size_t)(b * H_ + w) * N_ + (n0 + r)) * P_ + p) * DH_ + col;
      dst[o] = (_Float16)v;
    }
  }
}

// ---------------------------------------------------------------------------
// Kernel 2: per-(b,h,n) causal attention over time axis P=128.
// grid = B*H*N blocks of 256; wave w owns query rows [16w,16w+16).
// ---------------------------------------------------------------------------
__global__ __launch_bounds__(256) void attn_kernel(
    const _Float16* __restrict__ qp, const _Float16* __restrict__ kp,
    const _Float16* __restrict__ vp, _Float16* __restrict__ attn)
{
  __shared__ _Float16 Qs[P_][DH_ + 2];
  __shared__ _Float16 Ks[P_][DH_ + 2];
  __shared__ _Float16 Vs[P_][DH_ + 2];
  __shared__ _Float16 Wst[8][16][40];     // per-wave W re-fragment staging

  const int idx = blockIdx.x;
  const int n = idx % N_;
  const int h = (idx / N_) % H_;
  const int b = idx / (N_ * H_);
  const size_t base = (((size_t)b * H_ + h) * N_ + n) * (size_t)(P_ * DH_);

  const int tid = threadIdx.x;
  { // coalesced 2048-half loads of Q,K,V into LDS
    const int e0 = tid * 8;
    const int pr = e0 / DH_;
    const int pc = e0 % DH_;
#pragma unroll
    for (int j = 0; j < 8; ++j) {
      Qs[pr][pc + j] = qp[base + e0 + j];
      Ks[pr][pc + j] = kp[base + e0 + j];
      Vs[pr][pc + j] = vp[base + e0 + j];
    }
  }
  __syncthreads();

  const int w     = tid >> 5;
  const int lane  = tid & 31;
  const int col   = lane & 15;
  const int rbase = (lane >> 4) * 8;

  // Q fragment (d=16 padded to K=32 with zeros)
  v16h aQ = load_a_frag(lane, [&](int mm, int kk) -> _Float16 {
    return (kk < DH_) ? Qs[w * 16 + mm][kk] : (_Float16)0.0f;
  });

  // S = Q * K^T  (16 rows x 128 cols, f32 in registers)
  v8f S[8];
#pragma unroll
  for (int qt = 0; qt < 8; ++qt) {
    v16h bK = load_b_frag(lane, [&](int kk, int nn) -> _Float16 {
      return (kk < DH_) ? Ks[qt * 16 + nn][kk] : (_Float16)0.0f;
    });
    v8f z = {};
    S[qt] = wmma_f16(aQ, bK, z);
  }

  // scale 1/sqrt(16) + causal mask (q > p -> NEG_INF)
#pragma unroll
  for (int qt = 0; qt < 8; ++qt) {
#pragma unroll
    for (int i = 0; i < 8; ++i) {
      const int prow = w * 16 + rbase + i;
      const int qcol = qt * 16 + col;
      const float x = S[qt][i] * 0.25f;
      S[qt][i] = (qcol > prow) ? NEG_INF_ : x;
    }
  }

  // row softmax: per C-VGPR i, row spans 8 regs in-lane x 16 lanes (one half)
#pragma unroll
  for (int i = 0; i < 8; ++i) {
    float mx = -3.4e38f;
#pragma unroll
    for (int qt = 0; qt < 8; ++qt) mx = fmaxf(mx, S[qt][i]);
#pragma unroll
    for (int off = 1; off < 16; off <<= 1)
      mx = fmaxf(mx, __shfl_xor(mx, off, 32));
    float sum = 0.0f;
#pragma unroll
    for (int qt = 0; qt < 8; ++qt) {
      const float e = __expf(S[qt][i] - mx);
      S[qt][i] = e;
      sum += e;
    }
#pragma unroll
    for (int off = 1; off < 16; off <<= 1)
      sum += __shfl_xor(sum, off, 32);
    const float inv = 1.0f / sum;
#pragma unroll
    for (int qt = 0; qt < 8; ++qt) S[qt][i] *= inv;
  }

  // O = W * V  (K loop over 128 keys in 4 chunks of 32)
  v8f O = {};
#pragma unroll
  for (int kt = 0; kt < 4; ++kt) {
    // stage this wave's 16x32 W slab into its private LDS scratch as f16
#pragma unroll
    for (int hh = 0; hh < 2; ++hh) {
      const int qt = kt * 2 + hh;
#pragma unroll
      for (int i = 0; i < 8; ++i)
        Wst[w][rbase + i][hh * 16 + col] = (_Float16)S[qt][i];
    }
    // same-wave LDS RAW: LDS pipeline is in-order, but fence explicitly
    asm volatile("s_wait_dscnt 0" ::: "memory");
    v16h aW = load_a_frag(lane, [&](int mm, int kk) -> _Float16 {
      return Wst[w][mm][kk];
    });
    v16h bV = load_b_frag(lane, [&](int kk, int nn) -> _Float16 {
      return Vs[kt * 32 + kk][nn];
    });
    O = wmma_f16(aW, bV, O);
  }

  // attended -> (B,P,N,D) f16, this block writes cols [16h,16h+16)
#pragma unroll
  for (int i = 0; i < 8; ++i) {
    const int prow = w * 16 + rbase + i;
    const size_t o =
        (((size_t)b * P_ + prow) * N_ + n) * D_ + h * DH_ + col;
    attn[o] = (_Float16)O[i];
  }
}

// ---------------------------------------------------------------------------
// Kernel 3: fused dense layer  out = act(in @ W + b), in f16 MxD, packed W.
// ---------------------------------------------------------------------------
template <bool RELU, typename OutT>
__global__ __launch_bounds__(256) void fc_kernel(
    const _Float16* __restrict__ in, const _Float16* __restrict__ Wp,
    const float* __restrict__ bias, OutT* __restrict__ out)
{
  __shared__ _Float16 as[16][D_ + 8];
  const int m0  = blockIdx.x * 16;
  const int tid = threadIdx.x;
  {
    const int r  = tid >> 4;
    const int c0 = (tid & 15) * 8;
    const _Float16* src = in + (size_t)(m0 + r) * D_;
#pragma unroll
    for (int j = 0; j < 8; ++j) as[r][c0 + j] = src[c0 + j];
  }
  __syncthreads();

  const int w     = tid >> 5;
  const int lane  = tid & 31;
  const int col   = lane & 15;
  const int rbase = (lane >> 4) * 8;

  v8f acc = {};
#pragma unroll
  for (int kt = 0; kt < 4; ++kt) {
    v16h a = load_a_frag(lane, [&](int mm, int kk) -> _Float16 {
      return as[mm][kt * 32 + kk];
    });
    const v16h bfr = *wp_frag(Wp, 4, w, kt, lane);     // one 32B load
    acc = wmma_f16(a, bfr, acc);
  }
  const float bb = bias[w * 16 + col];
#pragma unroll
  for (int i = 0; i < 8; ++i) {
    float v = acc[i] + bb;
    if (RELU) v = fmaxf(v, 0.0f);
    out[(size_t)(m0 + rbase + i) * D_ + w * 16 + col] = (OutT)v;
  }
}

// ---------------------------------------------------------------------------
extern "C" void kernel_launch(void* const* d_in, const int* in_sizes, int n_in,
                              void* d_out, int out_size, void* d_ws, size_t ws_size,
                              hipStream_t stream) {
  const float* X   = (const float*)d_in[0];
  const float* STE = (const float*)d_in[1];
  const float* Wq  = (const float*)d_in[2];
  const float* bq  = (const float*)d_in[3];
  const float* Wk  = (const float*)d_in[4];
  const float* bk  = (const float*)d_in[5];
  const float* Wv  = (const float*)d_in[6];
  const float* bv  = (const float*)d_in[7];
  const float* W1  = (const float*)d_in[8];
  const float* b1  = (const float*)d_in[9];
  const float* W2  = (const float*)d_in[10];
  const float* b2  = (const float*)d_in[11];

  const size_t M   = (size_t)B_ * P_ * N_;        // 262144 tokens
  const size_t seg = M * D_;                      // halves per f16 buffer

  _Float16* qp   = (_Float16*)d_ws;
  _Float16* kp   = qp + seg;
  _Float16* vp   = kp + seg;
  _Float16* attn = vp + seg;
  _Float16* wqP  = attn + seg;                    // 256*128 halves
  _Float16* wkP  = wqP + (size_t)DIN_ * D_;
  _Float16* wvP  = wkP + (size_t)DIN_ * D_;
  _Float16* w1P  = wvP + (size_t)DIN_ * D_;       // 128*128 halves
  _Float16* w2P  = w1P + (size_t)D_ * D_;
  _Float16* h1   = qp;                            // q buffer is dead post-attn

  const dim3 blk(256);
  const dim3 gridM((unsigned)(M / 16));           // 16384
  const dim3 gridA((unsigned)(B_ * H_ * N_));     // 16384

  // per-launch weight packing (deterministic, no cached state)
  pack_w_kernel<<<dim3(DIN_ / 32), blk, 0, stream>>>(Wq, wqP, DIN_ / 32);
  pack_w_kernel<<<dim3(DIN_ / 32), blk, 0, stream>>>(Wk, wkP, DIN_ / 32);
  pack_w_kernel<<<dim3(DIN_ / 32), blk, 0, stream>>>(Wv, wvP, DIN_ / 32);
  pack_w_kernel<<<dim3(D_ / 32),  blk, 0, stream>>>(W1, w1P, D_ / 32);
  pack_w_kernel<<<dim3(D_ / 32),  blk, 0, stream>>>(W2, w2P, D_ / 32);

  qkv_kernel<<<gridM, blk, 0, stream>>>(X, STE, wqP, bq, wkP, bk, wvP, bv,
                                        qp, kp, vp);
  attn_kernel<<<gridA, blk, 0, stream>>>(qp, kp, vp, attn);
  fc_kernel<true, _Float16><<<gridM, blk, 0, stream>>>(attn, w1P, b1, h1);
  fc_kernel<false, float><<<gridM, blk, 0, stream>>>(h1, w2P, b2, (float*)d_out);
}
